// Decoder_39316130628122
// MI455X (gfx1250) — matrix operand order repaired and checked by reading, not compiled, not run on hardware
//
#include <hip/hip_runtime.h>
#include <hip/hip_bf16.h>

// ---------------------------------------------------------------- constants
#define T_ENC   168
#define BATCH   32
#define D_ENC   512
#define T_DEC   800
#define N_MELS  80
#define D_DEC   1024
#define D_PRE   256
#define D_ATT   128
#define N_LOC   32
#define LOC_K   31
#define N_FILT  512
#define PK      5

typedef __attribute__((ext_vector_type(16))) _Float16 v16h;
typedef __attribute__((ext_vector_type(8)))  _Float16 h8;
typedef __attribute__((ext_vector_type(8)))  float    v8f;

// output offsets (floats) in d_out: (mels_frames, decoder_frames, stop, align)
#define OUT_DEC   (BATCH * N_MELS * T_DEC)                 // 2,048,000
#define OUT_STOP  (2 * BATCH * N_MELS * T_DEC)             // 4,096,000
#define OUT_ALIGN (OUT_STOP + BATCH * T_DEC)               // 4,121,600

// workspace offsets (bytes), all 256-aligned
#define OFF_WCAT1  0UL                               // 4096*1792*2 = 14,680,064
#define OFF_WCAT2  14680064UL                        // 4096*2048*2 = 16,777,216
#define OFF_WQH    31457280UL                        // 128*1024*2  = 262,144
#define OFF_XALL   31719424UL                        // 25600*256*2 = 13,107,200
#define OFF_PM     44826624UL                        // 32*168*128*4= 2,752,512
#define OFF_STATE  47579136UL
#define   ST_H1    0UL
#define   ST_C1    131072UL
#define   ST_H2    262144UL
#define   ST_C2    393216UL
#define   ST_H1H   524288UL
#define   ST_H2H   589824UL
#define   ST_CTX   655360UL
#define   ST_CTXH  720896UL
#define   ST_CUM   753664UL
#define STATE_BYTES 775424UL
#define OFF_G      48354560UL                        // 32*4096*4 = 524,288
#define OFF_PQ     48878848UL                        // 32*128*4
#define OFF_E      48895232UL                        // 32*168*4 (+pad)
#define OFF_MEAN   48916992UL                        // 512*4
#define OFF_VAR    48919040UL
#define OFF_BUFA   48921088UL                        // 32*512*800*4 = 52,428,800 (aliases X1)
#define OFF_BUFB   101349888UL
// total ~153.8 MB

// ---------------------------------------------------------------- WMMA frags
__device__ inline v16h load_a_frag(const _Float16* __restrict__ A, int lda,
                                   int m0, int kk, int lane) {
    // 16-bit A 16x32 (ISA 7.12.2): lanes 0-15 row M=lane, K={0..7,16..23};
    // lanes 16-31 row M=lane-16, K={8..15,24..31}
    int row = m0 + (lane & 15);
    int kb  = kk + ((lane >> 4) << 3);
    const _Float16* p = A + (long)row * lda + kb;
    h8 lo = *(const h8*)(p);
    h8 hi = *(const h8*)(p + 16);
    v16h f;
#pragma unroll
    for (int i = 0; i < 8; ++i) { f[i] = lo[i]; f[8 + i] = hi[i]; }
    return f;
}

__device__ inline v16h load_b_frag(const _Float16* __restrict__ W, int ldw,
                                   int n0, int kk, int lane) {
    // 16-bit B 32x16: lanes 0-15 col N=lane K={0..7,16..23}; lanes 16-31 K={8..15,24..31}
    int col = n0 + (lane & 15);
    int kb  = kk + ((lane >> 4) << 3);
    const _Float16* p = W + (long)col * ldw + kb;
    h8 lo = *(const h8*)(p);
    h8 hi = *(const h8*)(p + 16);
    v16h f;
#pragma unroll
    for (int i = 0; i < 8; ++i) { f[i] = lo[i]; f[8 + i] = hi[i]; }
    return f;
}

// G[32,N] = A(32 x Ktot, split in up to 3 row-major f16 pieces) * W^T
// W is f16 row-major [N, Ktot]. One wave per 16-col tile; computes both
// M-tiles (batch=32) so each B fragment is reused twice.
__global__ __launch_bounds__(32) void wmma_gates(
    const _Float16* __restrict__ A0, int K0,
    const _Float16* __restrict__ A1, int K1,
    const _Float16* __restrict__ A2, int K2,
    const _Float16* __restrict__ W, int Ktot,
    float* __restrict__ G, int N)
{
    int lane = threadIdx.x;
    int n0 = blockIdx.x * 16;
    v8f acc0 = {}; v8f acc1 = {};
    for (int kk = 0; kk < Ktot; kk += 32) {
        const _Float16* A; int lda, kl;
        if (kk < K0)            { A = A0; lda = K0; kl = kk; }
        else if (kk < K0 + K1)  { A = A1; lda = K1; kl = kk - K0; }
        else                    { A = A2; lda = K2; kl = kk - K0 - K1; }
        v16h bf = load_b_frag(W, Ktot, n0, kk, lane);
        v16h a0 = load_a_frag(A, lda, 0, kl, lane);
        acc0 = __builtin_amdgcn_wmma_f32_16x16x32_f16(
            false, a0, false, bf, (short)0, acc0, false, false);
        v16h a1 = load_a_frag(A, lda, 16, kl, lane);
        acc1 = __builtin_amdgcn_wmma_f32_16x16x32_f16(
            false, a1, false, bf, (short)0, acc1, false, false);
    }
    // C/D layout: lane l -> col l&15, rows i + 8*(l>>4)
    int nc = n0 + (lane & 15);
    int mb = (lane >> 4) << 3;
#pragma unroll
    for (int i = 0; i < 8; ++i) {
        G[(long)(mb + i) * N + nc]      = acc0[i];
        G[(long)(16 + mb + i) * N + nc] = acc1[i];
    }
}

// ---------------------------------------------------------------- LSTM epilogue
__global__ void lstm_epilogue(const float* __restrict__ G,
                              const float* __restrict__ b_ih,
                              const float* __restrict__ b_hh,
                              float* __restrict__ c, float* __restrict__ h,
                              _Float16* __restrict__ hh)
{
    int idx = blockIdx.x * blockDim.x + threadIdx.x;
    if (idx >= BATCH * D_DEC) return;
    int b = idx >> 10, j = idx & 1023;
    const float* g = G + (long)b * 4 * D_DEC;
    float ig = g[j]        + b_ih[j]        + b_hh[j];
    float fg = g[j + 1024] + b_ih[j + 1024] + b_hh[j + 1024];
    float gg = g[j + 2048] + b_ih[j + 2048] + b_hh[j + 2048];
    float og = g[j + 3072] + b_ih[j + 3072] + b_hh[j + 3072];
    float si = 1.f / (1.f + __expf(-ig));
    float sf = 1.f / (1.f + __expf(-fg));
    float so = 1.f / (1.f + __expf(-og));
    float cn = sf * c[idx] + si * tanhf(gg);
    float hn = so * tanhf(cn);
    c[idx] = cn; h[idx] = hn; hh[idx] = (_Float16)hn;
}

// ---------------------------------------------------------------- attention
__global__ __launch_bounds__(128) void attn_energies(
    const float* __restrict__ cum, const float* __restrict__ pq,
    const float* __restrict__ pm,  const float* __restrict__ lock,
    const float* __restrict__ wl,  const float* __restrict__ av,
    const int* __restrict__ tlen,  float* __restrict__ e)
{
    int t = blockIdx.x, b = blockIdx.y, tid = threadIdx.x;
    __shared__ float loc[N_LOC];
    __shared__ float red[128];
    if (tid < N_LOC) {
        float s = 0.f;
        for (int k = 0; k < LOC_K; ++k) {
            int tt = t + k - (LOC_K - 1) / 2;
            float cv = (tt >= 0 && tt < T_ENC) ? cum[b * T_ENC + tt] : 0.f;
            s += lock[tid * LOC_K + k] * cv;
        }
        loc[tid] = s;
    }
    __syncthreads();
    float pl = 0.f;
#pragma unroll 8
    for (int f = 0; f < N_LOC; ++f) pl += wl[tid * N_LOC + f] * loc[f];
    float val = tanhf(pq[b * D_ATT + tid] +
                      pm[((long)b * T_ENC + t) * D_ATT + tid] + pl) * av[tid];
    red[tid] = val;
    __syncthreads();
    for (int s = 64; s > 0; s >>= 1) {
        if (tid < s) red[tid] += red[tid + s];
        __syncthreads();
    }
    if (tid == 0) e[b * T_ENC + t] = (t < tlen[b]) ? red[0] : -1e9f;
}

__global__ __launch_bounds__(256) void attn_softmax_ctx(
    const float* __restrict__ e, const float* __restrict__ enc,
    float* __restrict__ cum, float* __restrict__ ctx,
    _Float16* __restrict__ ctxh, float* __restrict__ out_align, int step)
{
    int b = blockIdx.x, tid = threadIdx.x;
    __shared__ float sa[T_ENC];
    __shared__ float red[256];
    float v = (tid < T_ENC) ? e[b * T_ENC + tid] : -3.4e38f;
    red[tid] = v; __syncthreads();
    for (int s = 128; s > 0; s >>= 1) {
        if (tid < s) red[tid] = fmaxf(red[tid], red[tid + s]);
        __syncthreads();
    }
    float m = red[0]; __syncthreads();
    float p = (tid < T_ENC) ? __expf(v - m) : 0.f;
    red[tid] = p; __syncthreads();
    for (int s = 128; s > 0; s >>= 1) {
        if (tid < s) red[tid] += red[tid + s];
        __syncthreads();
    }
    float inv = 1.f / red[0];
    if (tid < T_ENC) {
        float a = p * inv;
        sa[tid] = a;
        cum[b * T_ENC + tid] += a;
        out_align[((long)b * T_ENC + tid) * T_DEC + step] = a;
    }
    __syncthreads();
    for (int c = tid; c < D_ENC; c += 256) {
        float s = 0.f;
        for (int t = 0; t < T_ENC; ++t)
            s += sa[t] * enc[((long)t * BATCH + b) * D_ENC + c];
        ctx[b * D_ENC + c] = s;
        ctxh[b * D_ENC + c] = (_Float16)s;
    }
}

// ---------------------------------------------------------------- projections
__global__ void proj_kernel(const float* __restrict__ h2,
                            const float* __restrict__ ctx,
                            const float* __restrict__ mw,
                            const float* __restrict__ mb,
                            const float* __restrict__ sw,
                            const float* __restrict__ sb,
                            float* __restrict__ out, int step)
{
    int idx = blockIdx.x * blockDim.x + threadIdx.x;
    if (idx >= BATCH * (N_MELS + 1)) return;
    int b = idx / (N_MELS + 1), o = idx % (N_MELS + 1);
    const float* w = (o < N_MELS) ? (mw + (long)o * (D_DEC + D_ENC)) : sw;
    float s = (o < N_MELS) ? mb[o] : sb[0];
    const float* hp = h2 + (long)b * D_DEC;
    const float* cp = ctx + (long)b * D_ENC;
    for (int k = 0; k < D_DEC; ++k) s += hp[k] * w[k];
    for (int k = 0; k < D_ENC; ++k) s += cp[k] * w[D_DEC + k];
    if (o < N_MELS)
        out[OUT_DEC + ((long)b * N_MELS + o) * T_DEC + step] = s;
    else
        out[OUT_STOP + (long)b * T_DEC + step] = s;
}

// ---------------------------------------------------------------- phase-0 prep
__global__ void zero_f32(float* p, long n) {
    long i = blockIdx.x * 256L + threadIdx.x;
    if (i < n) p[i] = 0.f;
}

__global__ void concat2_f16(const float* __restrict__ w0, int k0,
                            const float* __restrict__ w1, int k1,
                            _Float16* __restrict__ dst, int rows)
{
    int ktot = k0 + k1;
    long tot = (long)rows * ktot;
    long idx = blockIdx.x * 256L + threadIdx.x;
    if (idx >= tot) return;
    long n = idx / ktot; int k = (int)(idx % ktot);
    float v = (k < k0) ? w0[n * k0 + k] : w1[n * k1 + (k - k0)];
    dst[idx] = (_Float16)v;
}

__global__ void cvt_f16(const float* __restrict__ src, _Float16* __restrict__ dst, long n) {
    long i = blockIdx.x * 256L + threadIdx.x;
    if (i < n) dst[i] = (_Float16)src[i];
}

// prenet layer 1 over all 800 teacher-forced frames (no recurrence)
__global__ void prenet1(const float* __restrict__ mels,
                        const float* __restrict__ w1, const float* __restrict__ b1,
                        float* __restrict__ x1)
{
    int idx = blockIdx.x * blockDim.x + threadIdx.x;   // 25600*256
    if (idx >= T_DEC * BATCH * D_PRE) return;
    int r = idx >> 8, j = idx & 255;
    int t = r >> 5, b = r & 31;
    float s = b1[j];
    if (t > 0) {
        const float* mp = mels + (long)b * N_MELS * T_DEC + (t - 1);
        const float* w = w1 + j * N_MELS;
        for (int m = 0; m < N_MELS; ++m) s += w[m] * mp[(long)m * T_DEC];
    }
    x1[idx] = fmaxf(s, 0.f);
}

__global__ void prenet2(const float* __restrict__ x1,
                        const float* __restrict__ w2, const float* __restrict__ b2,
                        _Float16* __restrict__ xall)
{
    int idx = blockIdx.x * blockDim.x + threadIdx.x;
    if (idx >= T_DEC * BATCH * D_PRE) return;
    int r = idx >> 8, j = idx & 255;
    const float* x = x1 + (long)r * D_PRE;
    const float* w = w2 + (long)j * D_PRE;
    float s = b2[j];
    for (int k = 0; k < D_PRE; ++k) s += w[k] * x[k];
    xall[idx] = (_Float16)fmaxf(s, 0.f);
}

__global__ void compute_pm(const float* __restrict__ enc,
                           const float* __restrict__ wm, float* __restrict__ pm)
{
    int idx = blockIdx.x * blockDim.x + threadIdx.x;   // 32*168*128
    if (idx >= BATCH * T_ENC * D_ATT) return;
    int a = idx & 127; int bt = idx >> 7;
    int b = bt / T_ENC, t = bt % T_ENC;
    const float* ep = enc + ((long)t * BATCH + b) * D_ENC;
    const float* w = wm + (long)a * D_ENC;
    float s = 0.f;
    for (int c = 0; c < D_ENC; ++c) s += ep[c] * w[c];
    pm[idx] = s;
}

// ---------------------------------------------------------------- postnet
__global__ void conv1d(const float* __restrict__ x, const float* __restrict__ w,
                       const float* __restrict__ bias,
                       const float* __restrict__ residual,
                       float* __restrict__ y, int Cin, int Cout)
{
    long tot = (long)BATCH * Cout * T_DEC;
    long idx = blockIdx.x * 256L + threadIdx.x;
    if (idx >= tot) return;
    int t  = (int)(idx % T_DEC);
    int co = (int)((idx / T_DEC) % Cout);
    int b  = (int)(idx / ((long)T_DEC * Cout));
    float s = bias[co];
    const float* wp = w + (long)co * Cin * PK;
    const float* xp = x + (long)b * Cin * T_DEC;
    int klo = (t < 2) ? (2 - t) : 0;
    int khi = (t > T_DEC - 3) ? (T_DEC + 2 - t) : PK;
    for (int ci = 0; ci < Cin; ++ci) {
        const float* xc = xp + (long)ci * T_DEC + t - 2;
        const float* wc = wp + ci * PK;
        for (int dk = klo; dk < khi; ++dk) s += wc[dk] * xc[dk];
    }
    if (residual) s += residual[idx];
    y[idx] = s;
}

__global__ __launch_bounds__(256) void bn_stats(const float* __restrict__ x,
                                                float* __restrict__ mean,
                                                float* __restrict__ var, int C)
{
    int c = blockIdx.x, tid = threadIdx.x;
    __shared__ float s1[256], s2[256];
    float a = 0.f, q = 0.f;
    for (int i = tid; i < BATCH * T_DEC; i += 256) {
        int b = i / T_DEC, t = i % T_DEC;
        float v = x[((long)b * C + c) * T_DEC + t];
        a += v; q += v * v;
    }
    s1[tid] = a; s2[tid] = q; __syncthreads();
    for (int s = 128; s > 0; s >>= 1) {
        if (tid < s) { s1[tid] += s1[tid + s]; s2[tid] += s2[tid + s]; }
        __syncthreads();
    }
    if (tid == 0) {
        float m = s1[0] / (float)(BATCH * T_DEC);
        mean[c] = m;
        var[c] = s2[0] / (float)(BATCH * T_DEC) - m * m;
    }
}

__global__ void bn_tanh(const float* __restrict__ x, const float* __restrict__ mean,
                        const float* __restrict__ var, const float* __restrict__ gamma,
                        const float* __restrict__ beta, float* __restrict__ y, int C)
{
    long tot = (long)BATCH * C * T_DEC;
    long idx = blockIdx.x * 256L + threadIdx.x;
    if (idx >= tot) return;
    int c = (int)((idx / T_DEC) % C);
    float v = gamma[c] * (x[idx] - mean[c]) * rsqrtf(var[c] + 1e-5f) + beta[c];
    y[idx] = tanhf(v);
}

// ---------------------------------------------------------------- launch
extern "C" void kernel_launch(void* const* d_in, const int* in_sizes, int n_in,
                              void* d_out, int out_size, void* d_ws, size_t ws_size,
                              hipStream_t stream) {
    (void)in_sizes; (void)n_in; (void)out_size; (void)ws_size;
    const float* enc      = (const float*)d_in[0];
    const int*   tlen     = (const int*)  d_in[1];
    const float* mels     = (const float*)d_in[2];
    const float* pw1      = (const float*)d_in[3];
    const float* pb1      = (const float*)d_in[4];
    const float* pw2      = (const float*)d_in[5];
    const float* pb2      = (const float*)d_in[6];
    const float* w_ih1    = (const float*)d_in[7];
    const float* w_hh1    = (const float*)d_in[8];
    const float* b_ih1    = (const float*)d_in[9];
    const float* b_hh1    = (const float*)d_in[10];
    const float* w_ih2    = (const float*)d_in[11];
    const float* w_hh2    = (const float*)d_in[12];
    const float* b_ih2    = (const float*)d_in[13];
    const float* b_hh2    = (const float*)d_in[14];
    const float* att_wq   = (const float*)d_in[15];
    const float* att_wm   = (const float*)d_in[16];
    const float* lock     = (const float*)d_in[17];
    const float* att_wl   = (const float*)d_in[18];
    const float* att_v    = (const float*)d_in[19];
    const float* mproj_w  = (const float*)d_in[20];
    const float* mproj_b  = (const float*)d_in[21];
    const float* sproj_w  = (const float*)d_in[22];
    const float* sproj_b  = (const float*)d_in[23];
    const float* cw[5]; const float* cb[5];
    const float* bng[4]; const float* bnb[4];
    for (int i = 0; i < 5; ++i) { cw[i] = (const float*)d_in[24 + i]; cb[i] = (const float*)d_in[29 + i]; }
    for (int i = 0; i < 4; ++i) { bng[i] = (const float*)d_in[34 + i]; bnb[i] = (const float*)d_in[38 + i]; }

    char* ws = (char*)d_ws;
    _Float16* WCAT1 = (_Float16*)(ws + OFF_WCAT1);
    _Float16* WCAT2 = (_Float16*)(ws + OFF_WCAT2);
    _Float16* WQH   = (_Float16*)(ws + OFF_WQH);
    _Float16* XALL  = (_Float16*)(ws + OFF_XALL);
    float*    PM    = (float*)(ws + OFF_PM);
    char*     st    = ws + OFF_STATE;
    float*    H1    = (float*)(st + ST_H1);
    float*    C1    = (float*)(st + ST_C1);
    float*    H2    = (float*)(st + ST_H2);
    float*    C2    = (float*)(st + ST_C2);
    _Float16* H1H   = (_Float16*)(st + ST_H1H);
    _Float16* H2H   = (_Float16*)(st + ST_H2H);
    float*    CTX   = (float*)(st + ST_CTX);
    _Float16* CTXH  = (_Float16*)(st + ST_CTXH);
    float*    CUM   = (float*)(st + ST_CUM);
    float*    G     = (float*)(ws + OFF_G);
    float*    PQ    = (float*)(ws + OFF_PQ);
    float*    E     = (float*)(ws + OFF_E);
    float*    MEAN  = (float*)(ws + OFF_MEAN);
    float*    VAR   = (float*)(ws + OFF_VAR);
    float*    BUFA  = (float*)(ws + OFF_BUFA);   // also prenet x1 staging
    float*    BUFB  = (float*)(ws + OFF_BUFB);
    float*    out   = (float*)d_out;

    // ---- phase 0: state init, f16 weight prep, prenet-for-all-t, pm
    long stw = STATE_BYTES / 4;
    zero_f32<<<(int)((stw + 255) / 256), 256, 0, stream>>>((float*)st, stw);
    {
        long tot1 = 4096L * 1792;
        concat2_f16<<<(int)((tot1 + 255) / 256), 256, 0, stream>>>(w_ih1, 768, w_hh1, 1024, WCAT1, 4096);
        long tot2 = 4096L * 2048;
        concat2_f16<<<(int)((tot2 + 255) / 256), 256, 0, stream>>>(w_ih2, 1024, w_hh2, 1024, WCAT2, 4096);
        cvt_f16<<<(128 * 1024 + 255) / 256, 256, 0, stream>>>(att_wq, WQH, 128L * 1024);
    }
    {
        int tot = T_DEC * BATCH * D_PRE;  // 6,553,600
        prenet1<<<(tot + 255) / 256, 256, 0, stream>>>(mels, pw1, pb1, BUFA);
        prenet2<<<(tot + 255) / 256, 256, 0, stream>>>(BUFA, pw2, pb2, XALL);
    }
    {
        int tot = BATCH * T_ENC * D_ATT;
        compute_pm<<<(tot + 255) / 256, 256, 0, stream>>>(enc, att_wm, PM);
    }

    // ---- sequential attention-LSTM scan (weights stay resident in 192MB L2)
    for (int t = 0; t < T_DEC; ++t) {
        const _Float16* xt = XALL + (long)t * BATCH * D_PRE;
        // LSTM1 gates: A = [prenet_x | ctx | h1], K = 256+512+1024 = 1792
        wmma_gates<<<4 * D_DEC / 16, 32, 0, stream>>>(
            xt, D_PRE, CTXH, D_ENC, H1H, D_DEC, WCAT1, 1792, G, 4 * D_DEC);
        lstm_epilogue<<<(BATCH * D_DEC + 255) / 256, 256, 0, stream>>>(
            G, b_ih1, b_hh1, C1, H1, H1H);
        // LSTM2 gates: A = [h1 | h2], K = 2048
        wmma_gates<<<4 * D_DEC / 16, 32, 0, stream>>>(
            H1H, D_DEC, H2H, D_DEC, H2H, 0, WCAT2, 2048, G, 4 * D_DEC);
        lstm_epilogue<<<(BATCH * D_DEC + 255) / 256, 256, 0, stream>>>(
            G, b_ih2, b_hh2, C2, H2, H2H);
        // attention query: pq = h2 @ wq^T   [32,128]
        wmma_gates<<<D_ATT / 16, 32, 0, stream>>>(
            H2H, D_DEC, H2H, 0, H2H, 0, WQH, D_DEC, PQ, D_ATT);
        attn_energies<<<dim3(T_ENC, BATCH), 128, 0, stream>>>(
            CUM, PQ, PM, lock, att_wl, att_v, tlen, E);
        attn_softmax_ctx<<<BATCH, 256, 0, stream>>>(
            E, enc, CUM, CTX, CTXH, out + OUT_ALIGN, t);
        proj_kernel<<<(BATCH * (N_MELS + 1) + 255) / 256, 256, 0, stream>>>(
            H2, CTX, mproj_w, mproj_b, sproj_w, sproj_b, out, t);
    }

    // ---- postnet: 4x (conv -> BN(batch stats) -> tanh), final conv + residual
    const float* dec = out + OUT_DEC;
    long totF = (long)BATCH * N_FILT * T_DEC;      // 13,107,200
    int gF = (int)((totF + 255) / 256);
    conv1d<<<gF, 256, 0, stream>>>(dec, cw[0], cb[0], nullptr, BUFA, N_MELS, N_FILT);
    bn_stats<<<N_FILT, 256, 0, stream>>>(BUFA, MEAN, VAR, N_FILT);
    bn_tanh<<<gF, 256, 0, stream>>>(BUFA, MEAN, VAR, bng[0], bnb[0], BUFB, N_FILT);
    for (int l = 1; l < 4; ++l) {
        conv1d<<<gF, 256, 0, stream>>>(BUFB, cw[l], cb[l], nullptr, BUFA, N_FILT, N_FILT);
        bn_stats<<<N_FILT, 256, 0, stream>>>(BUFA, MEAN, VAR, N_FILT);
        bn_tanh<<<gF, 256, 0, stream>>>(BUFA, MEAN, VAR, bng[l], bnb[l], BUFB, N_FILT);
    }
    long totM = (long)BATCH * N_MELS * T_DEC;
    conv1d<<<(int)((totM + 255) / 256), 256, 0, stream>>>(
        BUFB, cw[4], cb[4], dec, out, N_FILT, N_MELS);
}
